// GPTLanguageModel_16569983828375
// MI455X (gfx1250) — compile-verified, matrix operand order checked by reading, not run on hardware
//
#include <hip/hip_runtime.h>
#include <hip/hip_bf16.h>
#include <math.h>

// ---------------- model dims ----------------
#define B_   64
#define T_   256
#define C_   384
#define H_   6
#define HS_  64
#define L_   6
#define V_   96
#define FF_  1536
#define BT_  (B_*T_)
#define EPS_ 1e-5f

typedef __bf16 bf16_t;
typedef __bf16 v16bf __attribute__((ext_vector_type(16)));
typedef float  v8f   __attribute__((ext_vector_type(8)));

__device__ __forceinline__ bf16_t f2bf(float f) {
  unsigned u = __builtin_bit_cast(unsigned, f);
  unsigned r = (u + 0x7FFFu + ((u >> 16) & 1u)) >> 16;   // round-to-nearest-even
  return __builtin_bit_cast(bf16_t, (unsigned short)r);
}

// two contiguous 16B loads per lane -> one 16x32 bf16 operand fragment
// (lanes 0-15: K 0..7 & 16..23 ; lanes 16-31: K 8..15 & 24..31)
__device__ __forceinline__ v16bf load_frag(const bf16_t* p) {
  v16bf f;
  ((uint4*)&f)[0] = *(const uint4*)(p);
  ((uint4*)&f)[1] = *(const uint4*)(p + 16);
  return f;
}

// Async Global->LDS DMA (CDNA5): writes LDS directly, no VGPR staging,
// tracked with ASYNCcnt.  lds_addr = LDS byte address (low 32 bits of a
// generic pointer into __shared__ memory, per the LDS aperture rule).
__device__ __forceinline__ void async_load_b128(unsigned lds_addr, const void* gptr) {
  asm volatile("global_load_async_to_lds_b128 %0, %1, off"
               :: "v"(lds_addr), "v"(gptr) : "memory");
}
__device__ __forceinline__ void wait_async0() {
  asm volatile("s_wait_asynccnt 0x0" ::: "memory");
}

// ---------------------------------------------------------------------------
// Batched WMMA GEMM, async-LDS staged + double-buffered:
//   block = 256 threads (8 waves, 4(M) x 2(N)), block tile 128 x 64, k-step 32.
//   Each k-step: the block stages A(128x32) and B(64x32) bf16 tiles into LDS
//   using global_load_async_to_lds_b128 (3 per thread, ASYNCcnt-tracked,
//   zero VGPR staging), issued for the *next* buffer before this step's
//   WMMAs so the DMA overlaps the matrix ops; s_wait_asynccnt 0 + barrier
//   closes the step.  Waves pull operand fragments with ds_load_b128 and
//   issue 4 v_wmma_f32_16x16x32_bf16 each (2x2 register blocking, 32x32
//   output per wave).  LDS rows padded to 40 elements -> conflict-free reads.
// A is [M][K] bf16 row-major; Bt is [N][K] bf16 (weights pre-transposed).
// Requires M % 128 == 0; N handled with row clamp + store guard (N % 32 == 0).
// ---------------------------------------------------------------------------
enum { EPI_F32 = 0, EPI_BF16 = 1, EPI_F32_BIAS_RES = 2, EPI_BF16_BIAS_RELU = 3, EPI_F32_BIAS = 4 };

template<int EPI>
__global__ __launch_bounds__(256) void wmma_gemm(
    const bf16_t* __restrict__ A,  int lda, long sAb, long sAh,
    const bf16_t* __restrict__ Bt, int ldb, long sBb, long sBh,
    void* __restrict__ Dv,         int ldd, long sDb, long sDh,
    const float* __restrict__ bias, int M, int N, int K, int nH)
{
  constexpr int LDSROW = 40;              // padded row stride (elements)
  constexpr int AOFFE  = 0;               // A tile: rows 0..127
  constexpr int BOFFE  = 128 * LDSROW;    // B tile: rows 0..63
  constexpr int BUFE   = 192 * LDSROW;    // elements per buffer
  __shared__ __align__(16) bf16_t smem[2][BUFE];   // 2 x 15360 B = 30 KB

  const int z  = blockIdx.z;
  const int bi = z / nH, hi = z - bi * nH;
  A  += (long)bi * sAb + (long)hi * sAh;
  Bt += (long)bi * sBb + (long)hi * sBh;
  const long dbase = (long)bi * sDb + (long)hi * sDh;

  const int tid  = threadIdx.x;
  const int w    = tid >> 5;
  const int lane = tid & 31;
  const int m0blk = blockIdx.y * 128;
  const int n0blk = blockIdx.x * 64;
  const int wmOff = (w >> 1) * 32;        // wave M offset in block tile
  const int wnOff = (w & 1)  * 32;        // wave N offset in block tile

  // cooperative staging assignment: each thread owns three 16B chunks
  const int arow0 = tid >> 2;             // 0..63
  const int aoff  = (tid & 3) * 8;        // 0,8,16,24 (elements)
  const int arow1 = arow0 + 64;
  const int brow  = arow0;
  const int browg = (n0blk + brow < N) ? (n0blk + brow) : (N - 1);  // clamp (lm head)

  const bf16_t* aptr0 = A  + (long)(m0blk + arow0) * lda + aoff;
  const bf16_t* aptr1 = A  + (long)(m0blk + arow1) * lda + aoff;
  const bf16_t* bptr  = Bt + (long)browg * ldb + aoff;

  // LDS byte addresses of this thread's three staging chunks (buffer 0)
  const unsigned ldsBase = (unsigned)(unsigned long long)&smem[0][0];
  const unsigned bufB    = (unsigned)(BUFE * sizeof(bf16_t));
  const unsigned la0 = ldsBase + (unsigned)((AOFFE + arow0 * LDSROW + aoff) * sizeof(bf16_t));
  const unsigned la1 = ldsBase + (unsigned)((AOFFE + arow1 * LDSROW + aoff) * sizeof(bf16_t));
  const unsigned lb  = ldsBase + (unsigned)((BOFFE + brow  * LDSROW + aoff) * sizeof(bf16_t));

  // prologue: async-stage k-step 0 into buffer 0
  async_load_b128(la0, aptr0);
  async_load_b128(la1, aptr1);
  async_load_b128(lb,  bptr);
  wait_async0();
  __syncthreads();

  const int rr = lane & 15;               // row (A) / col (B) owned by this lane
  const int kb = (lane >> 4) << 3;        // 0 or 8: lane-group K base

  v8f a00 = {0.f,0.f,0.f,0.f,0.f,0.f,0.f,0.f};
  v8f a01 = a00, a10 = a00, a11 = a00;

  const int nsteps = K >> 5;
  for (int s = 0; s < nsteps; ++s) {
    const int buf = s & 1;
    if (s + 1 < nsteps) {                 // DMA next step into the other buffer
      const int g = (s + 1) << 5;
      const unsigned nb = (unsigned)(buf ^ 1) * bufB;
      async_load_b128(la0 + nb, aptr0 + g);
      async_load_b128(la1 + nb, aptr1 + g);
      async_load_b128(lb  + nb, bptr  + g);
    }
    const bf16_t* As = &smem[buf][AOFFE];
    const bf16_t* Bs = &smem[buf][BOFFE];
    const v16bf fa0 = load_frag(As + (wmOff      + rr) * LDSROW + kb);
    const v16bf fa1 = load_frag(As + (wmOff + 16 + rr) * LDSROW + kb);
    const v16bf fb0 = load_frag(Bs + (wnOff      + rr) * LDSROW + kb);
    const v16bf fb1 = load_frag(Bs + (wnOff + 16 + rr) * LDSROW + kb);
    a00 = __builtin_amdgcn_wmma_f32_16x16x32_bf16(false, fa0, false, fb0, (short)0, a00, false, false);
    a01 = __builtin_amdgcn_wmma_f32_16x16x32_bf16(false, fa0, false, fb1, (short)0, a01, false, false);
    a10 = __builtin_amdgcn_wmma_f32_16x16x32_bf16(false, fa1, false, fb0, (short)0, a10, false, false);
    a11 = __builtin_amdgcn_wmma_f32_16x16x32_bf16(false, fa1, false, fb1, (short)0, a11, false, false);
    if (s + 1 < nsteps) {                 // close the step: DMA done + barrier
      wait_async0();
      __syncthreads();
    }
  }

  // C/D layout: VGPR i -> M = i (+8 for lanes 16-31), N = lane & 15
  const int col = lane & 15;
  const int rbse = (lane >> 4) << 3;
  const int m0 = m0blk + wmOff;
  const int n0 = n0blk + wnOff;
  v8f accs[2][2] = {{a00, a01}, {a10, a11}};
  #pragma unroll
  for (int ti = 0; ti < 2; ++ti) {
    #pragma unroll
    for (int tj = 0; tj < 2; ++tj) {
      const int nn = n0 + tj * 16 + col;
      if (nn < N) {
        #pragma unroll
        for (int i = 0; i < 8; ++i) {
          const int  row = m0 + ti * 16 + rbse + i;
          const long off = dbase + (long)row * ldd + nn;
          const float v = accs[ti][tj][i];
          if constexpr (EPI == EPI_F32) {
            ((float*)Dv)[off] = v;
          } else if constexpr (EPI == EPI_BF16) {
            ((bf16_t*)Dv)[off] = f2bf(v);
          } else if constexpr (EPI == EPI_F32_BIAS_RES) {
            float* D = (float*)Dv;
            D[off] = D[off] + v + bias[nn];
          } else if constexpr (EPI == EPI_BF16_BIAS_RELU) {
            const float t = v + bias[nn];
            ((bf16_t*)Dv)[off] = f2bf(t > 0.f ? t : 0.f);
          } else { // EPI_F32_BIAS
            ((float*)Dv)[off] = v + bias[nn];
          }
        }
      }
    }
  }
}

// ---------------- weight transpose + f32 -> bf16 convert -------------------
// src: [K][N] f32 row-major  ->  dst: [N][K] bf16
__global__ __launch_bounds__(256) void transpose_f32_to_bf16(
    const float* __restrict__ src, bf16_t* __restrict__ dst, int K, int N)
{
  __shared__ float tile[32][33];
  const int n0 = blockIdx.x * 32, k0 = blockIdx.y * 32;
  const int tx = threadIdx.x, ty = threadIdx.y;
  for (int i = ty; i < 32; i += 8) {
    const int k = k0 + i, n = n0 + tx;
    tile[i][tx] = (k < K && n < N) ? src[(long)k * N + n] : 0.f;
  }
  __syncthreads();
  for (int i = ty; i < 32; i += 8) {
    const int n = n0 + i, k = k0 + tx;
    if (n < N && k < K) dst[(long)n * K + k] = f2bf(tile[tx][i]);
  }
}

// v [b][t][h*HS+d] (bf16) -> vt [b][h][d][t] (bf16)
__global__ __launch_bounds__(256) void transpose_v_kernel(
    const bf16_t* __restrict__ v, bf16_t* __restrict__ vt)
{
  const long i = (long)blockIdx.x * blockDim.x + threadIdx.x;   // over B*H*HS*T
  const int s = (int)(i % T_);
  long r = i / T_;
  const int d = (int)(r % HS_); r /= HS_;
  const int h = (int)(r % H_);
  const long b = r / H_;
  vt[i] = v[(b * T_ + s) * (long)C_ + h * HS_ + d];
}

// ---------------- embedding ----------------
__global__ __launch_bounds__(128) void embed_kernel(
    const int* __restrict__ idx, const float* __restrict__ tok,
    const float* __restrict__ pos, float* __restrict__ x)
{
  const int row = blockIdx.x;          // b*T + t
  const int t   = row & (T_ - 1);
  const long to = (long)idx[row] * C_;
  for (int c = threadIdx.x; c < C_; c += 128)
    x[(long)row * C_ + c] = tok[to + c] + pos[(long)t * C_ + c];
}

// ---------------- layernorm (f32 in, bf16 out) ----------------
__global__ __launch_bounds__(128) void layernorm_bf16(
    const float* __restrict__ x, const float* __restrict__ g,
    const float* __restrict__ b, bf16_t* __restrict__ out)
{
  const int row = blockIdx.x, tid = threadIdx.x;
  const float* xr = x + (long)row * C_;
  __shared__ float red[128];

  float s = 0.f;
  for (int c = tid; c < C_; c += 128) s += xr[c];
  red[tid] = s; __syncthreads();
  for (int o = 64; o > 0; o >>= 1) { if (tid < o) red[tid] += red[tid + o]; __syncthreads(); }
  const float mean = red[0] / (float)C_;
  __syncthreads();

  float s2 = 0.f;
  for (int c = tid; c < C_; c += 128) { const float d = xr[c] - mean; s2 += d * d; }
  red[tid] = s2; __syncthreads();
  for (int o = 64; o > 0; o >>= 1) { if (tid < o) red[tid] += red[tid + o]; __syncthreads(); }
  const float rstd = rsqrtf(red[0] / (float)C_ + EPS_);

  for (int c = tid; c < C_; c += 128)
    out[(long)row * C_ + c] = f2bf((xr[c] - mean) * rstd * g[c] + b[c]);
}

// ---------------- causal softmax: S f32 -> P bf16 ----------------
__global__ __launch_bounds__(256) void softmax_causal(
    const float* __restrict__ S, bf16_t* __restrict__ P, float scale)
{
  const int t  = blockIdx.x;
  const long bh = blockIdx.y;
  const long ro = (bh * T_ + t) * (long)T_;
  const int s = threadIdx.x;
  __shared__ float red[256];

  const float v = (s <= t) ? S[ro + s] * scale : -3.402823466e38f;
  red[s] = v; __syncthreads();
  for (int o = 128; o > 0; o >>= 1) { if (s < o) red[s] = fmaxf(red[s], red[s + o]); __syncthreads(); }
  const float mx = red[0]; __syncthreads();

  const float e = (s <= t) ? __expf(v - mx) : 0.f;
  red[s] = e; __syncthreads();
  for (int o = 128; o > 0; o >>= 1) { if (s < o) red[s] += red[s + o]; __syncthreads(); }
  const float inv = 1.f / red[0];

  P[ro + s] = f2bf(e * inv);
}

// ---------------- per-row log-softmax loss ----------------
__global__ __launch_bounds__(128) void row_logloss(
    const float* __restrict__ logits, const int* __restrict__ targets,
    float* __restrict__ rowloss)
{
  const int row = blockIdx.x, tid = threadIdx.x;
  const float* lr = logits + (long)row * V_;
  __shared__ float red[128];

  const float v = (tid < V_) ? lr[tid] : -3.402823466e38f;
  red[tid] = v; __syncthreads();
  for (int o = 64; o > 0; o >>= 1) { if (tid < o) red[tid] = fmaxf(red[tid], red[tid + o]); __syncthreads(); }
  const float mx = red[0]; __syncthreads();

  red[tid] = (tid < V_) ? __expf(v - mx) : 0.f; __syncthreads();
  for (int o = 64; o > 0; o >>= 1) { if (tid < o) red[tid] += red[tid + o]; __syncthreads(); }
  if (tid == 0) rowloss[row] = logf(red[0]) + mx - lr[targets[row]];
}

__global__ __launch_bounds__(256) void reduce_mean(
    const float* __restrict__ rl, int n, float* __restrict__ out)
{
  __shared__ float red[256];
  const int tid = threadIdx.x;
  float s = 0.f;
  for (int i = tid; i < n; i += 256) s += rl[i];
  red[tid] = s; __syncthreads();
  for (int o = 128; o > 0; o >>= 1) { if (tid < o) red[tid] += red[tid + o]; __syncthreads(); }
  if (tid == 0) *out = red[0] / (float)n;
}

// ===========================================================================
extern "C" void kernel_launch(void* const* d_in, const int* in_sizes, int n_in,
                              void* d_out, int out_size, void* d_ws, size_t ws_size,
                              hipStream_t stream) {
  const int*   idx     = (const int*)  d_in[0];
  const int*   targets = (const int*)  d_in[1];
  const float* tok_emb = (const float*)d_in[2];
  const float* pos_emb = (const float*)d_in[3];
  const float* Wq  = (const float*)d_in[4];
  const float* Wk  = (const float*)d_in[5];
  const float* Wv  = (const float*)d_in[6];
  const float* Wo  = (const float*)d_in[7];
  const float* bo  = (const float*)d_in[8];
  const float* W1  = (const float*)d_in[9];
  const float* b1  = (const float*)d_in[10];
  const float* W2  = (const float*)d_in[11];
  const float* b2  = (const float*)d_in[12];
  const float* ln1g = (const float*)d_in[13];
  const float* ln1b = (const float*)d_in[14];
  const float* ln2g = (const float*)d_in[15];
  const float* ln2b = (const float*)d_in[16];
  const float* lnfg = (const float*)d_in[17];
  const float* lnfb = (const float*)d_in[18];
  const float* lm_W = (const float*)d_in[19];
  const float* lm_b = (const float*)d_in[20];

  float* logits = (float*)d_out;
  float* lossp  = logits + (size_t)BT_ * V_;

  // ---- workspace bump allocator (256B aligned) ----
  char* base = (char*)d_ws; size_t used = 0;
  auto alloc = [&](size_t bytes) -> void* {
    void* r = base + used;
    used = (used + bytes + 255) & ~(size_t)255;
    return r;
  };
  bf16_t* WqT  = (bf16_t*)alloc((size_t)L_ * C_ * C_ * 2);
  bf16_t* WkT  = (bf16_t*)alloc((size_t)L_ * C_ * C_ * 2);
  bf16_t* WvT  = (bf16_t*)alloc((size_t)L_ * C_ * C_ * 2);
  bf16_t* WoT  = (bf16_t*)alloc((size_t)L_ * C_ * C_ * 2);
  bf16_t* W1T  = (bf16_t*)alloc((size_t)L_ * FF_ * C_ * 2);
  bf16_t* W2T  = (bf16_t*)alloc((size_t)L_ * C_ * FF_ * 2);
  bf16_t* lmWT = (bf16_t*)alloc((size_t)V_ * C_ * 2);
  float*  x    = (float*) alloc((size_t)BT_ * C_ * 4);
  bf16_t* hbuf = (bf16_t*)alloc((size_t)BT_ * C_ * 2);
  bf16_t* qbuf = (bf16_t*)alloc((size_t)BT_ * C_ * 2);
  bf16_t* kbuf = (bf16_t*)alloc((size_t)BT_ * C_ * 2);
  bf16_t* vbuf = (bf16_t*)alloc((size_t)BT_ * C_ * 2);
  bf16_t* vt   = (bf16_t*)alloc((size_t)BT_ * C_ * 2);
  bf16_t* attn = (bf16_t*)alloc((size_t)BT_ * C_ * 2);
  const int BCH = 8;                                   // attention batch chunk
  float*  Sbuf = (float*) alloc((size_t)BCH * H_ * T_ * T_ * 4);
  bf16_t* Pbuf = (bf16_t*)alloc((size_t)BCH * H_ * T_ * T_ * 2);
  const int MCH = 4096;                                // FFN row chunk
  bf16_t* mid  = (bf16_t*)alloc((size_t)MCH * FF_ * 2);
  float*  rowl = (float*) alloc((size_t)BT_ * 4);
  if (used > ws_size) return;

  const dim3 tb32x8(32, 8);

  // ---- weights: transpose + convert to bf16 ([N][K] layouts) ----
  for (int l = 0; l < L_; ++l) {
    for (int h = 0; h < H_; ++h) {
      const size_t so = ((size_t)(l * H_ + h)) * C_ * HS_;         // [C][HS] block
      const size_t dofs = ((size_t)l * C_ + (size_t)h * HS_) * C_; // -> [n=h*64+d][C]
      transpose_f32_to_bf16<<<dim3(HS_/32, C_/32), tb32x8, 0, stream>>>(Wq + so, WqT + dofs, C_, HS_);
      transpose_f32_to_bf16<<<dim3(HS_/32, C_/32), tb32x8, 0, stream>>>(Wk + so, WkT + dofs, C_, HS_);
      transpose_f32_to_bf16<<<dim3(HS_/32, C_/32), tb32x8, 0, stream>>>(Wv + so, WvT + dofs, C_, HS_);
    }
    transpose_f32_to_bf16<<<dim3(C_/32,  C_/32 ), tb32x8, 0, stream>>>(Wo + (size_t)l*C_*C_,  WoT + (size_t)l*C_*C_,  C_,  C_);
    transpose_f32_to_bf16<<<dim3(FF_/32, C_/32 ), tb32x8, 0, stream>>>(W1 + (size_t)l*C_*FF_, W1T + (size_t)l*FF_*C_, C_,  FF_);
    transpose_f32_to_bf16<<<dim3(C_/32,  FF_/32), tb32x8, 0, stream>>>(W2 + (size_t)l*FF_*C_, W2T + (size_t)l*C_*FF_, FF_, C_);
  }
  transpose_f32_to_bf16<<<dim3(V_/32, C_/32), tb32x8, 0, stream>>>(lm_W, lmWT, C_, V_);

  // ---- embeddings ----
  embed_kernel<<<BT_, 128, 0, stream>>>(idx, tok_emb, pos_emb, x);

  const float attn_scale = 1.f / sqrtf((float)C_);   // reference scales by C^-0.5

  for (int l = 0; l < L_; ++l) {
    // h = LN1(x)
    layernorm_bf16<<<BT_, 128, 0, stream>>>(x, ln1g + (size_t)l*C_, ln1b + (size_t)l*C_, hbuf);

    // q,k,v = h @ W{q,k,v}   [BT x 384]
    wmma_gemm<EPI_BF16><<<dim3(C_/64, BT_/128, 1), 256, 0, stream>>>(
        hbuf, C_, 0, 0, WqT + (size_t)l*C_*C_, C_, 0, 0, qbuf, C_, 0, 0, nullptr, BT_, C_, C_, 1);
    wmma_gemm<EPI_BF16><<<dim3(C_/64, BT_/128, 1), 256, 0, stream>>>(
        hbuf, C_, 0, 0, WkT + (size_t)l*C_*C_, C_, 0, 0, kbuf, C_, 0, 0, nullptr, BT_, C_, C_, 1);
    wmma_gemm<EPI_BF16><<<dim3(C_/64, BT_/128, 1), 256, 0, stream>>>(
        hbuf, C_, 0, 0, WvT + (size_t)l*C_*C_, C_, 0, 0, vbuf, C_, 0, 0, nullptr, BT_, C_, C_, 1);

    // vt[b][h][d][s] = v[b][s][h][d]
    transpose_v_kernel<<<(BT_*(long)C_)/256, 256, 0, stream>>>(vbuf, vt);

    // attention, 8 batches at a time
    for (int bc = 0; bc < B_/BCH; ++bc) {
      const size_t qo = (size_t)bc * BCH * T_ * C_;
      // S[b',h] = q @ k^T   (M=N=256, K=64)
      wmma_gemm<EPI_F32><<<dim3(T_/64, T_/128, BCH*H_), 256, 0, stream>>>(
          qbuf + qo, C_, (long)T_*C_, HS_,
          kbuf + qo, C_, (long)T_*C_, HS_,
          Sbuf, T_, (long)H_*T_*T_, (long)T_*T_, nullptr, T_, T_, HS_, H_);
      // P = softmax(mask(S * scale))
      softmax_causal<<<dim3(T_, BCH*H_), 256, 0, stream>>>(Sbuf, Pbuf, attn_scale);
      // attn[b',h] = P @ V   (M=256, N=64, K=256)
      wmma_gemm<EPI_BF16><<<dim3(HS_/64, T_/128, BCH*H_), 256, 0, stream>>>(
          Pbuf, T_, (long)H_*T_*T_, (long)T_*T_,
          vt + (size_t)bc*BCH*C_*T_, T_, (long)C_*T_, (long)HS_*T_,
          attn + qo, C_, (long)T_*C_, HS_, nullptr, T_, HS_, T_, H_);
    }

    // x += attn @ Wo + bo
    wmma_gemm<EPI_F32_BIAS_RES><<<dim3(C_/64, BT_/128, 1), 256, 0, stream>>>(
        attn, C_, 0, 0, WoT + (size_t)l*C_*C_, C_, 0, 0,
        x, C_, 0, 0, bo + (size_t)l*C_, BT_, C_, C_, 1);

    // h2 = LN2(x)
    layernorm_bf16<<<BT_, 128, 0, stream>>>(x, ln2g + (size_t)l*C_, ln2b + (size_t)l*C_, hbuf);

    // FFN, 4096 rows at a time:  x += relu(h2 @ W1 + b1) @ W2 + b2
    for (int mc = 0; mc < BT_/MCH; ++mc) {
      const size_t ro = (size_t)mc * MCH;
      wmma_gemm<EPI_BF16_BIAS_RELU><<<dim3(FF_/64, MCH/128, 1), 256, 0, stream>>>(
          hbuf + ro*C_, C_, 0, 0, W1T + (size_t)l*FF_*C_, C_, 0, 0,
          mid, FF_, 0, 0, b1 + (size_t)l*FF_, MCH, FF_, C_, 1);
      wmma_gemm<EPI_F32_BIAS_RES><<<dim3(C_/64, MCH/128, 1), 256, 0, stream>>>(
          mid, FF_, 0, 0, W2T + (size_t)l*C_*FF_, FF_, 0, 0,
          x + ro*C_, C_, 0, 0, b2 + (size_t)l*C_, MCH, C_, FF_, 1);
    }
  }

  // final LN + lm_head
  layernorm_bf16<<<BT_, 128, 0, stream>>>(x, lnfg, lnfb, hbuf);
  wmma_gemm<EPI_F32_BIAS><<<dim3((V_+63)/64, BT_/128, 1), 256, 0, stream>>>(
      hbuf, C_, 0, 0, lmWT, C_, 0, 0, logits, V_, 0, 0, lm_b, BT_, V_, C_, 1);

  // loss = mean(logsumexp(logits) - logits[target])
  row_logloss<<<BT_, 128, 0, stream>>>(logits, targets, rowl);
  reduce_mean<<<1, 256, 0, stream>>>(rowl, BT_, lossp);
}
